// AttentionModel_26723286516244
// MI455X (gfx1250) — matrix operand order repaired and checked by reading, not compiled, not run on hardware
//
// MI455X / gfx1250 full-pipeline implementation (round 4).
// Heavy GEMMs: k_gemm_fast — 16x64 strip per wave, 4x v_wmma_f32_16x16x32_bf16
// per K-step, float4 clause loads, 32-bit saddr offsets.  Transposed-W and
// masked paths use two-phase load-then-convert so strided gathers clause up.
// Convs are templated on kernel size to kill runtime div/mod.  Fused selective
// scan keeps state in registers and B/C in LDS (saves ~4.3 GB of HBM traffic).

#include <hip/hip_runtime.h>
#include <hip/hip_bf16.h>
#include <math.h>

typedef __bf16 bf16_t;
typedef bf16_t v16bf __attribute__((ext_vector_type(16)));
typedef float  v8f   __attribute__((ext_vector_type(8)));

#define ACT_NONE     0
#define ACT_GELU     1
#define ACT_SILU     2
#define ACT_SIGMOID  3
#define ACT_SOFTPLUS 4

__device__ __forceinline__ float act_apply(float v, int act) {
  switch (act) {
    case ACT_GELU:     return 0.5f * v * (1.0f + erff(v * 0.70710678118f));
    case ACT_SILU:     return v / (1.0f + __expf(-v));
    case ACT_SIGMOID:  return 1.0f / (1.0f + __expf(-v));
    case ACT_SOFTPLUS: return (v > 20.0f) ? v : log1pf(__expf(v));
    default:           return v;
  }
}

__device__ __forceinline__ v8f wmma_bf16(v16bf a, v16bf b, v8f c) {
  return __builtin_amdgcn_wmma_f32_16x16x32_bf16(false, a, false, b, (short)0, c,
                                                 false, false);
}
__device__ __forceinline__ int iclamp(int v, int lo, int hi) {
  return v < lo ? lo : (v > hi ? hi : v);
}

// Fragment maps (wave32):
//  A (16x32 bf16): row M = lane&15 for both halves; k(i) = k0 + hs*16 + i
//  B (32x16 bf16): col N = lane&15;                 k(i) = k0 + hs*16 + i
//  C (16x16 f32):  M = j + 8*hs, N = lane&15

__device__ __forceinline__ void cvt16(v16bf& dst, const float4& x0, const float4& x1,
                                      const float4& x2, const float4& x3) {
  dst[0] = (bf16_t)x0.x;  dst[1] = (bf16_t)x0.y;  dst[2] = (bf16_t)x0.z;  dst[3] = (bf16_t)x0.w;
  dst[4] = (bf16_t)x1.x;  dst[5] = (bf16_t)x1.y;  dst[6] = (bf16_t)x1.z;  dst[7] = (bf16_t)x1.w;
  dst[8] = (bf16_t)x2.x;  dst[9] = (bf16_t)x2.y;  dst[10] = (bf16_t)x2.z; dst[11] = (bf16_t)x2.w;
  dst[12] = (bf16_t)x3.x; dst[13] = (bf16_t)x3.y; dst[14] = (bf16_t)x3.z; dst[15] = (bf16_t)x3.w;
}
__device__ __forceinline__ void cvt16a(v16bf& dst, const float* t) {
#pragma unroll
  for (int i = 0; i < 16; ++i) dst[i] = (bf16_t)t[i];
}

// ---------------------------------------------------------------------------
// Fast GEMM: requires M%16==0, N%64==0, K%32==0 (checked on host).
// C = act(scale * A @ op(W) + bias) + res ;  one wave = 16x64 strip.
// ---------------------------------------------------------------------------
__global__ void k_gemm_fast(const float* __restrict__ A, int lda,
                            const float* __restrict__ Wt, int ldw, int wtrans,
                            const float* __restrict__ bias,
                            const float* __restrict__ res,
                            float* __restrict__ C, int ldc,
                            int K, float scale, int act,
                            int tilesM, int tilesN4, int zdiv,
                            long long aSH, long long aSL,
                            long long wSH, long long wSL,
                            long long cSH, long long cSL) {
  int wave = threadIdx.x >> 5, lane = threadIdx.x & 31;
  int tile = blockIdx.x * 8 + wave;
  if (tile >= tilesM * tilesN4) return;
  int tm = tile / tilesN4, tn4 = tile % tilesN4;
  int z = blockIdx.z, zq = z / zdiv, zr = z % zdiv;
  const float* Ab = A + zq * aSH + zr * aSL;
  const float* Wb = Wt + zq * wSH + zr * wSL;
  float*       Cb = C + zq * cSH + zr * cSL;
  const float* Rb = res ? (res + zq * cSH + zr * cSL) : (const float*)0;

  int hs = lane >> 4, rc = lane & 15;
  int mA = tm * 16 + rc;
  int nbase = tn4 * 64;
  v8f acc[4];
#pragma unroll
  for (int u = 0; u < 4; ++u) acc[u] = v8f{0.f, 0.f, 0.f, 0.f, 0.f, 0.f, 0.f, 0.f};

  const float* ap = Ab + mA * lda;
  int woff[4];
#pragma unroll
  for (int u = 0; u < 4; ++u)
    woff[u] = wtrans ? (nbase + u * 16 + rc) : (nbase + u * 16 + rc) * ldw;

  for (int k0 = 0; k0 < K; k0 += 32) {
    __builtin_prefetch(ap + k0 + 128, 0, 1);
    const float4* a4  = (const float4*)(ap + k0 + hs * 16);
    const float4* a4b = (const float4*)(ap + k0 + hs * 16 + 8);
    v16bf af;
    cvt16(af, a4[0], a4[1], a4b[0], a4b[1]);
#pragma unroll
    for (int u = 0; u < 4; ++u) {
      v16bf bf;
      if (!wtrans) {
        const float4* w4 = (const float4*)(Wb + woff[u] + k0 + hs * 16);
        cvt16(bf, w4[0], w4[1], w4[2], w4[3]);
      } else {
        int base = woff[u] + (k0 + hs * 16) * ldw;
        float t[16];
#pragma unroll
        for (int i = 0; i < 16; ++i) t[i] = Wb[base + i * ldw];  // one clause
        cvt16a(bf, t);
      }
      acc[u] = wmma_bf16(af, bf, acc[u]);
    }
  }
#pragma unroll
  for (int u = 0; u < 4; ++u) {
    int n = nbase + u * 16 + rc;
    float bv = bias ? bias[n] : 0.0f;
#pragma unroll
    for (int j = 0; j < 8; ++j) {
      int m = tm * 16 + j + 8 * hs;
      float v = act_apply(acc[u][j] * scale + bv, act);
      if (Rb) v += Rb[m * ldc + n];
      Cb[m * ldc + n] = v;
    }
  }
}

// ---------------------------------------------------------------------------
// Small/edge GEMM: one masked 16x16 tile per wave (coord-attention + final FC).
// ---------------------------------------------------------------------------
__global__ void k_gemm_small(const float* __restrict__ A, int lda,
                             const float* __restrict__ Wt, int ldw, int wtrans,
                             const float* __restrict__ bias,
                             const float* __restrict__ res,
                             float* __restrict__ C, int ldc,
                             int M, int N, int K, float scale, int act,
                             int tilesM, int tilesN, int zdiv,
                             long long aSH, long long aSL,
                             long long wSH, long long wSL,
                             long long cSH, long long cSL) {
  int wave = threadIdx.x >> 5, lane = threadIdx.x & 31;
  int tile = blockIdx.x * 8 + wave;
  if (tile >= tilesM * tilesN) return;
  int tm = tile / tilesN, tn = tile % tilesN;
  int z = blockIdx.z, zq = z / zdiv, zr = z % zdiv;
  const float* Ab = A + zq * aSH + zr * aSL;
  const float* Wb = Wt + zq * wSH + zr * wSL;
  float*       Cb = C + zq * cSH + zr * cSL;
  const float* Rb = res ? (res + zq * cSH + zr * cSL) : (const float*)0;

  int hs = lane >> 4, rc = lane & 15;
  int mA = tm * 16 + rc, nB = tn * 16 + rc;
  int mAc = iclamp(mA, 0, M - 1), nBc = iclamp(nB, 0, N - 1);
  float am = (mA < M) ? 1.0f : 0.0f;
  float wm = (nB < N) ? 1.0f : 0.0f;
  v8f acc = {0.f, 0.f, 0.f, 0.f, 0.f, 0.f, 0.f, 0.f};
  for (int k0 = 0; k0 < K; k0 += 32) {
    float ta[16], tb[16], ma[16], mb[16];
#pragma unroll
    for (int i = 0; i < 16; ++i) {
      int ka = k0 + hs * 16 + i;
      int kc = iclamp(ka, 0, K - 1);
      ma[i] = (ka < K) ? am : 0.0f;
      ta[i] = Ab[mAc * lda + kc];
      int kb = k0 + hs * 16 + i;
      int kbc = iclamp(kb, 0, K - 1);
      mb[i] = (kb < K) ? wm : 0.0f;
      tb[i] = wtrans ? Wb[kbc * ldw + nBc] : Wb[nBc * ldw + kbc];
    }
    v16bf af, bfr;
#pragma unroll
    for (int i = 0; i < 16; ++i) {
      af[i]  = (bf16_t)(ta[i] * ma[i]);
      bfr[i] = (bf16_t)(tb[i] * mb[i]);
    }
    acc = wmma_bf16(af, bfr, acc);
  }
#pragma unroll
  for (int j = 0; j < 8; ++j) {
    int m = tm * 16 + j + 8 * hs;
    int n = tn * 16 + rc;
    if (m < M && n < N) {
      float v = acc[j] * scale;
      if (bias) v += bias[n];
      v = act_apply(v, act);
      if (Rb) v += Rb[m * ldc + n];
      Cb[m * ldc + n] = v;
    }
  }
}

// ---------------------------------------------------------------------------
// Conv1d (NCH, OIH weights, 'SAME') as implicit-im2col WMMA GEMM.
// Templated on kernel size so k/KS, k%KS fold to constant ops.
// ---------------------------------------------------------------------------
template <int KS>
__global__ void k_conv_nch(const float* __restrict__ X, const float* __restrict__ Wt,
                           const float* __restrict__ bias, float* __restrict__ Out,
                           int inC, int T, int OC, int ocBase, int OCtot,
                           int tilesM, int tilesN) {
  int wave = threadIdx.x >> 5, lane = threadIdx.x & 31;
  int tile = blockIdx.x * 8 + wave;
  if (tile >= tilesM * tilesN) return;
  int tm = tile / tilesN, tn = tile % tilesN;
  int b = blockIdx.z;
  int hs = lane >> 4, rc = lane & 15;
  const int K = inC * KS, pad = KS >> 1;
  int t = tm * 16 + rc, n = tn * 16 + rc;
  int nc = iclamp(n, 0, OC - 1);
  float wmsk = (n < OC) ? 1.0f : 0.0f;
  const float* Xb = X + b * inC * T;
  v8f acc = {0.f, 0.f, 0.f, 0.f, 0.f, 0.f, 0.f, 0.f};
  for (int k0 = 0; k0 < K; k0 += 32) {
    float ta[16], tb[16], ma[16], mb[16];
#pragma unroll
    for (int i = 0; i < 16; ++i) {
      int ka = k0 + hs * 16 + i;
      int kc = iclamp(ka, 0, K - 1);
      int ic = kc / KS, tap = kc % KS;     // constant divisor
      int tt = t + tap - pad;
      ma[i] = (ka < K && tt >= 0 && tt < T) ? 1.0f : 0.0f;
      int ttc = iclamp(tt, 0, T - 1);
      ta[i] = Xb[ic * T + ttc];
      int kb = k0 + hs * 16 + i;
      int kbc = iclamp(kb, 0, K - 1);
      mb[i] = (kb < K) ? wmsk : 0.0f;
      tb[i] = Wt[nc * K + kbc];
    }
    v16bf af, bfr;
#pragma unroll
    for (int i = 0; i < 16; ++i) {
      af[i]  = (bf16_t)(ta[i] * ma[i]);
      bfr[i] = (bf16_t)(tb[i] * mb[i]);
    }
    acc = wmma_bf16(af, bfr, acc);
  }
  float* Ob = Out + b * OCtot * T;
#pragma unroll
  for (int j = 0; j < 8; ++j) {
    int m = tm * 16 + j + 8 * hs;
    int nn = tn * 16 + rc;
    if (m < T && nn < OC) {
      float v = act_apply(acc[j] + bias[nn], ACT_GELU);
      Ob[(ocBase + nn) * T + m] = v;
    }
  }
}

// ---------------------------------------------------------------------------
// Conv over L, ks=3, channel-last ('SAME'), HWIO weights (1,3,512,512).
// Din fixed at 512 -> tap = k>>9, ic = k&511.
// ---------------------------------------------------------------------------
#define DTRND 512
__global__ void k_conv_chlast(const float* __restrict__ X, const float* __restrict__ Wt,
                              const float* __restrict__ bias, float* __restrict__ Out,
                              int L, int tilesM, int tilesN) {
  int wave = threadIdx.x >> 5, lane = threadIdx.x & 31;
  int tile = blockIdx.x * 8 + wave;
  if (tile >= tilesM * tilesN) return;
  int tm = tile / tilesN, tn = tile % tilesN;
  int b = blockIdx.z;
  int hs = lane >> 4, rc = lane & 15;
  const int K = 3 * DTRND;
  int l = tm * 16 + rc, n = tn * 16 + rc;
  const float* Xb = X + b * L * DTRND;
  v8f acc = {0.f, 0.f, 0.f, 0.f, 0.f, 0.f, 0.f, 0.f};
  for (int k0 = 0; k0 < K; k0 += 32) {
    float ta[16], tb[16], ma[16];
#pragma unroll
    for (int i = 0; i < 16; ++i) {
      int ka = k0 + hs * 16 + i;
      int kc = iclamp(ka, 0, K - 1);
      int tap = kc >> 9, ic = kc & (DTRND - 1);
      int ll = l + tap - 1;
      ma[i] = (ka < K && ll >= 0 && ll < L) ? 1.0f : 0.0f;
      int llc = iclamp(ll, 0, L - 1);
      ta[i] = Xb[llc * DTRND + ic];
      int kb = k0 + hs * 16 + i;
      int kbc = iclamp(kb, 0, K - 1);
      tb[i] = (kb < K) ? Wt[kbc * DTRND + n] : 0.0f;
    }
    v16bf af, bfr;
#pragma unroll
    for (int i = 0; i < 16; ++i) {
      af[i]  = (bf16_t)(ta[i] * ma[i]);
      bfr[i] = (bf16_t)tb[i];
    }
    acc = wmma_bf16(af, bfr, acc);
  }
#pragma unroll
  for (int j = 0; j < 8; ++j) {
    int m = tm * 16 + j + 8 * hs;
    int nn = tn * 16 + rc;
    if (m < L && nn < DTRND)
      Out[(b * L + m) * DTRND + nn] = acc[j] + bias[nn];
  }
}

// ------------------------- element-wise / small kernels ---------------------
__global__ void k_perm(const float* __restrict__ src, float* __restrict__ dst,
                       int B, int D1, int D2) {
  int idx = blockIdx.x * 256 + threadIdx.x;
  int total = B * D1 * D2;
  if (idx >= total) return;
  int b = idx / (D1 * D2);
  int r = idx % (D1 * D2);
  int d1 = r / D2, d2 = r % D2;
  dst[(b * D2 + d2) * D1 + d1] = src[idx];
}

__global__ void k_maxpool3(const float* __restrict__ X, float* __restrict__ Out,
                           int B, int Cc, int T) {
  int idx = blockIdx.x * 256 + threadIdx.x;
  int total = B * Cc * T;
  if (idx >= total) return;
  int t = idx % T;
  int base = idx - t;
  float m = X[idx];
  if (t > 0)     m = fmaxf(m, X[base + t - 1]);
  if (t < T - 1) m = fmaxf(m, X[base + t + 1]);
  Out[idx] = m;
}

__global__ void k_flip(const float* __restrict__ X, float* __restrict__ Out,
                       int B, int L, int Cc) {
  int idx = blockIdx.x * 256 + threadIdx.x;
  int total = B * L * Cc;
  if (idx >= total) return;
  int c = idx % Cc;
  int l = (idx / Cc) % L;
  int b = idx / (Cc * L);
  Out[(b * L + (L - 1 - l)) * Cc + c] = X[idx];
}

__global__ void k_copy2(const float* __restrict__ src, float* __restrict__ dst,
                        int rows, int cols, int sld, int dld) {
  int idx = blockIdx.x * 256 + threadIdx.x;
  if (idx >= rows * cols) return;
  int r = idx / cols;
  int c = idx % cols;
  dst[r * dld + c] = src[r * sld + c];
}

__global__ void k_layernorm(const float* __restrict__ X, const float* __restrict__ g,
                            const float* __restrict__ bta, float* __restrict__ Y, int Cc) {
  __shared__ float red[256];
  int row = blockIdx.x;
  const float* xr = X + row * Cc;
  float s = 0.f;
  for (int c = threadIdx.x; c < Cc; c += 256) s += xr[c];
  red[threadIdx.x] = s; __syncthreads();
  for (int o = 128; o > 0; o >>= 1) {
    if (threadIdx.x < o) red[threadIdx.x] += red[threadIdx.x + o];
    __syncthreads();
  }
  float mu = red[0] / Cc; __syncthreads();
  float v = 0.f;
  for (int c = threadIdx.x; c < Cc; c += 256) { float d = xr[c] - mu; v += d * d; }
  red[threadIdx.x] = v; __syncthreads();
  for (int o = 128; o > 0; o >>= 1) {
    if (threadIdx.x < o) red[threadIdx.x] += red[threadIdx.x + o];
    __syncthreads();
  }
  float rstd = rsqrtf(red[0] / Cc + 1e-5f);
  for (int c = threadIdx.x; c < Cc; c += 256)
    Y[row * Cc + c] = (xr[c] - mu) * rstd * g[c] + bta[c];
}

__global__ void k_softmax(float* __restrict__ S, int L) {
  __shared__ float red[256];
  float* row = S + blockIdx.x * L;
  float mx = -3.4e38f;
  for (int c = threadIdx.x; c < L; c += 256) mx = fmaxf(mx, row[c]);
  red[threadIdx.x] = mx; __syncthreads();
  for (int o = 128; o > 0; o >>= 1) {
    if (threadIdx.x < o) red[threadIdx.x] = fmaxf(red[threadIdx.x], red[threadIdx.x + o]);
    __syncthreads();
  }
  float m = red[0]; __syncthreads();
  float s = 0.f;
  for (int c = threadIdx.x; c < L; c += 256) {
    float e = __expf(row[c] - m); row[c] = e; s += e;
  }
  red[threadIdx.x] = s; __syncthreads();
  for (int o = 128; o > 0; o >>= 1) {
    if (threadIdx.x < o) red[threadIdx.x] += red[threadIdx.x + o];
    __syncthreads();
  }
  float inv = 1.0f / red[0];
  for (int c = threadIdx.x; c < L; c += 256) row[c] *= inv;
}

// Depthwise causal conv (DCONV=4) + bias + SiLU, reading xi from xz[...,:DI].
#define DI_ 1024
#define DS_ 16
__global__ void k_dwconv_silu(const float* __restrict__ xz, const float* __restrict__ w,
                              const float* __restrict__ bias, float* __restrict__ xi,
                              int B, int L) {
  int idx = blockIdx.x * 256 + threadIdx.x;
  int total = B * L * DI_;
  if (idx >= total) return;
  int d = idx % DI_;
  int l = (idx / DI_) % L;
  int b = idx / (DI_ * L);
  int base = b * L * 2 * DI_;
  float acc = bias[d];
#pragma unroll
  for (int j = 0; j < 4; ++j) {
    int ll = l - 3 + j;
    if (ll >= 0) acc += w[d * 4 + j] * xz[base + ll * 2 * DI_ + d];
  }
  xi[idx] = acc * (1.0f / (1.0f + __expf(-acc)));   // SiLU
}

// Fused selective scan: states in registers, B/C staged in LDS, gated output
// written in-place over xi.  One thread per channel d, grid (DI/256, B).
__global__ void k_scan(const float* __restrict__ xz, float* __restrict__ xi,
                       const float* __restrict__ dt, const float* __restrict__ xdb,
                       const float* __restrict__ A_log, const float* __restrict__ Dv,
                       int L) {
  __shared__ float sB[32][DS_], sC[32][DS_];
  int tid = threadIdx.x;
  int d = blockIdx.x * 256 + tid;
  int b = blockIdx.y;
  float Ad[DS_], h[DS_];
#pragma unroll
  for (int s = 0; s < DS_; ++s) { Ad[s] = -__expf(A_log[d * DS_ + s]); h[s] = 0.f; }
  float Dd = Dv[d];
  for (int l0 = 0; l0 < L; l0 += 32) {
    __syncthreads();
    for (int j = tid; j < 32 * DS_; j += 256) {
      int li = j >> 4, s = j & 15;
      int base = (b * L + l0 + li) * 64;
      sB[li][s] = xdb[base + 32 + s];
      sC[li][s] = xdb[base + 48 + s];
    }
    __syncthreads();
    for (int li = 0; li < 32; ++li) {
      int l = l0 + li;
      int ix = (b * L + l) * DI_ + d;
      float xiv = xi[ix];
      float dtv = dt[ix];
      float zv  = xz[(b * L + l) * 2 * DI_ + DI_ + d];
      float acc = 0.f;
#pragma unroll
      for (int s = 0; s < DS_; ++s) {
        float a = __expf(dtv * Ad[s]);
        h[s] = a * h[s] + dtv * sB[li][s] * xiv;
        acc += h[s] * sC[li][s];
      }
      float yv = acc + xiv * Dd;
      yv *= zv * (1.0f / (1.0f + __expf(-zv)));   // * silu(z)
      xi[ix] = yv;
    }
  }
}

// coord attention helpers
__global__ void k_mean_t(const float* __restrict__ X, float* __restrict__ ycat,
                         int B, int T, int Cc) {
  int idx = blockIdx.x * 256 + threadIdx.x;
  if (idx >= B * Cc) return;
  int c = idx % Cc, b = idx / Cc;
  float s = 0.f;
  for (int t = 0; t < T; ++t) s += X[(b * T + t) * Cc + c];
  ycat[(b * (T + 1) + T) * Cc + c] = s / T;
}

__global__ void k_ycat_copy(const float* __restrict__ X, float* __restrict__ ycat,
                            int B, int T, int Cc) {
  int idx = blockIdx.x * 256 + threadIdx.x;
  if (idx >= B * T * Cc) return;
  int c = idx % Cc;
  int t = (idx / Cc) % T;
  int b = idx / (Cc * T);
  ycat[(b * (T + 1) + t) * Cc + c] = X[idx];
}

__global__ void k_bn_hswish(float* __restrict__ y, const float* __restrict__ mean,
                            const float* __restrict__ var, const float* __restrict__ g,
                            const float* __restrict__ bta, int total, int Cc) {
  int idx = blockIdx.x * 256 + threadIdx.x;
  if (idx >= total) return;
  int c = idx % Cc;
  float v = y[idx];
  v = (v - mean[c]) * rsqrtf(var[c] + 1e-5f) * g[c] + bta[c];
  float r6 = fminf(fmaxf(v + 3.0f, 0.0f), 6.0f);
  y[idx] = v * r6 * (1.0f / 6.0f);
}

__global__ void k_coord_mul(const float* __restrict__ X, const float* __restrict__ ah,
                            const float* __restrict__ aw, float* __restrict__ Out,
                            int B, int T, int Cc) {
  int idx = blockIdx.x * 256 + threadIdx.x;
  if (idx >= B * T * Cc) return;
  int c = idx % Cc;
  int b = idx / (Cc * T);
  Out[idx] = X[idx] * ah[idx] * aw[b * Cc + c];
}

// ------------------------------ host side -----------------------------------
static const int Bc = 16, Tc = 512, Fc = 512, INc = 9;

static inline void gemm(hipStream_t s, const float* A, int lda,
                        const float* Wt, int ldw, int wtrans,
                        const float* bias, const float* res, float* C, int ldc,
                        int M, int N, int K, float scale, int act,
                        int nz, int zdiv, long long aSH, long long aSL,
                        long long wSH, long long wSL, long long cSH, long long cSL) {
  if ((M % 16 == 0) && (N % 64 == 0) && (K % 32 == 0)) {
    int tilesM = M / 16, tilesN4 = N / 64;
    dim3 g((unsigned)((tilesM * tilesN4 + 7) / 8), 1, (unsigned)nz), blk(256, 1, 1);
    k_gemm_fast<<<g, blk, 0, s>>>(A, lda, Wt, ldw, wtrans, bias, res, C, ldc, K,
                                  scale, act, tilesM, tilesN4, zdiv,
                                  aSH, aSL, wSH, wSL, cSH, cSL);
  } else {
    int tilesM = (M + 15) / 16, tilesN = (N + 15) / 16;
    dim3 g((unsigned)((tilesM * tilesN + 7) / 8), 1, (unsigned)nz), blk(256, 1, 1);
    k_gemm_small<<<g, blk, 0, s>>>(A, lda, Wt, ldw, wtrans, bias, res, C, ldc,
                                   M, N, K, scale, act, tilesM, tilesN, zdiv,
                                   aSH, aSL, wSH, wSL, cSH, cSL);
  }
}
static inline void gemm1(hipStream_t s, const float* A, int lda,
                         const float* Wt, int ldw, const float* bias,
                         const float* res, float* C, int ldc,
                         int M, int N, int K, int act) {
  gemm(s, A, lda, Wt, ldw, 0, bias, res, C, ldc, M, N, K, 1.0f, act,
       1, 1, 0, 0, 0, 0, 0, 0);
}
static inline void conv_nch(hipStream_t s, const float* X, int inC, int ks,
                            const float* Wt, const float* bias, float* Out,
                            int ocBase, int OCtot, int OC) {
  int tilesM = Tc / 16, tilesN = (OC + 15) / 16;
  dim3 g((unsigned)((tilesM * tilesN + 7) / 8), 1, (unsigned)Bc), blk(256, 1, 1);
  if (ks == 1)
    k_conv_nch<1><<<g, blk, 0, s>>>(X, Wt, bias, Out, inC, Tc, OC, ocBase, OCtot, tilesM, tilesN);
  else if (ks == 3)
    k_conv_nch<3><<<g, blk, 0, s>>>(X, Wt, bias, Out, inC, Tc, OC, ocBase, OCtot, tilesM, tilesN);
  else
    k_conv_nch<5><<<g, blk, 0, s>>>(X, Wt, bias, Out, inC, Tc, OC, ocBase, OCtot, tilesM, tilesN);
}
#define NBLK(total) ((unsigned)(((long long)(total) + 255) / 256))

static void run_mamba(hipStream_t s, const float* xin, const float* const* p,
                      float* out, float* xz, float* xi, float* xdb, float* dtb) {
  const int M = Bc * Tc;
  gemm1(s, xin, 512, p[0], 512, 0, 0, xz, 2048, M, 2048, 512, ACT_NONE);
  k_dwconv_silu<<<NBLK((long long)M * DI_), 256, 0, s>>>(xz, p[1], p[2], xi, Bc, Tc);
  gemm1(s, xi, 1024, p[3], 1024, 0, 0, xdb, 64, M, 64, 1024, ACT_NONE);
  gemm1(s, xdb, 64, p[4], 32, p[5], 0, dtb, 1024, M, 1024, 32, ACT_SOFTPLUS);
  k_scan<<<dim3(4, Bc), 256, 0, s>>>(xz, xi, dtb, xdb, p[6], p[7], Tc);
  gemm1(s, xi, 1024, p[8], 1024, 0, 0, out, 512, M, 512, 1024, ACT_NONE);
}

extern "C" void kernel_launch(void* const* d_in, const int* in_sizes, int n_in,
                              void* d_out, int out_size, void* d_ws, size_t ws_size,
                              hipStream_t stream) {
  (void)in_sizes; (void)n_in; (void)out_size; (void)ws_size;
  #define P(i) ((const float*)d_in[i])
  float* W0 = (float*)d_ws;
  const long long SZ1 = (long long)Bc * Tc * Fc;           // 4.19M floats
  float* A0 = W0;                 // 4*SZ1 : xz / concat / inception outs
  float* A1 = A0 + 4 * SZ1;       // 2*SZ1 : xi / inc2 pool
  float* A2 = A1 + 2 * SZ1;       // 2*SZ1 : dt / Q,K / inception tbr
  float* A3 = A2 + 2 * SZ1;       // SZ1   : xn / ycat / qc,kc / o2 / X3
  float* A4 = A3 + SZ1;           // SZ1   : X (persistent residual)
  float* A5 = A4 + SZ1;           // SZ1   : f / a_h / V
  float* A6 = A5 + SZ1;           // SZ1   : b-tmp / attention O
  float* A7 = A6 + SZ1;           // SZ1   : flip tmp / coord out (X2)
  float* A8 = A7 + SZ1;           // SZ1   : misc
  float* xdb    = A8;                                     // 16*512*64
  float* scores = xdb + (long long)Bc * Tc * 64;          // 8*512*512
  float* y1     = scores + (long long)8 * Tc * Tc;        // 16*513*16
  float* awb    = y1 + (long long)Bc * 513 * 16;          // 16*512
  float* xnch   = awb + (long long)Bc * Fc;               // 16*9*512
  float* pool1  = xnch + (long long)Bc * INc * Tc;        // 16*9*512

  // ---- inception 1 & 2 -----------------------------------------------------
  k_perm<<<NBLK((long long)Bc * Tc * INc), 256, 0, stream>>>(P(0), xnch, Bc, Tc, INc);
  {
    const float* pr[12]; for (int j = 0; j < 12; ++j) pr[j] = P(1 + j);
    float* o = A0;
    conv_nch(stream, xnch, INc, 1, pr[0], pr[1], o, 0, 512, 128);
    conv_nch(stream, xnch, INc, 1, pr[2], pr[3], A2, 0, 128, 128);
    conv_nch(stream, A2, 128, 3, pr[4], pr[5], o, 128, 512, 128);
    conv_nch(stream, xnch, INc, 1, pr[6], pr[7], A2, 0, 128, 128);
    conv_nch(stream, A2, 128, 5, pr[8], pr[9], o, 256, 512, 128);
    k_maxpool3<<<NBLK((long long)Bc * INc * Tc), 256, 0, stream>>>(xnch, pool1, Bc, INc, Tc);
    conv_nch(stream, pool1, INc, 1, pr[10], pr[11], o, 384, 512, 128);
  }
  {
    const float* pr[12]; for (int j = 0; j < 12; ++j) pr[j] = P(13 + j);
    const float* xin = A0; float* o = A0 + SZ1;
    conv_nch(stream, xin, 512, 1, pr[0], pr[1], o, 0, 512, 128);
    conv_nch(stream, xin, 512, 1, pr[2], pr[3], A2, 0, 128, 128);
    conv_nch(stream, A2, 128, 3, pr[4], pr[5], o, 128, 512, 128);
    conv_nch(stream, xin, 512, 1, pr[6], pr[7], A2, 0, 128, 128);
    conv_nch(stream, A2, 128, 5, pr[8], pr[9], o, 256, 512, 128);
    k_maxpool3<<<NBLK(SZ1), 256, 0, stream>>>(xin, A1, Bc, 512, Tc);
    conv_nch(stream, A1, 512, 1, pr[10], pr[11], o, 384, 512, 128);
  }
  k_perm<<<NBLK(SZ1), 256, 0, stream>>>(A0 + SZ1, A4, Bc, 512, Tc);

  // ---- 2x bidirectional mamba (residual in-place on A4) --------------------
  for (int bi = 0; bi < 2; ++bi) {
    int base = (bi == 0) ? 25 : 47;
    const float* pf[9]; const float* pb[9];
    for (int j = 0; j < 9; ++j) { pf[j] = P(base + 2 + j); pb[j] = P(base + 11 + j); }
    k_layernorm<<<Bc * Tc, 256, 0, stream>>>(A4, P(base + 0), P(base + 1), A3, 512);
    run_mamba(stream, A3, pf, A5, A0, A1, xdb, A2);
    k_flip<<<NBLK(SZ1), 256, 0, stream>>>(A3, A7, Bc, Tc, 512);
    run_mamba(stream, A7, pb, A6, A0, A1, xdb, A2);
    k_flip<<<NBLK(SZ1), 256, 0, stream>>>(A6, A7, Bc, Tc, 512);
    k_copy2<<<NBLK(SZ1), 256, 0, stream>>>(A5, A0, Bc * Tc, 512, 512, 1024);
    k_copy2<<<NBLK(SZ1), 256, 0, stream>>>(A7, A0 + 512, Bc * Tc, 512, 512, 1024);
    gemm1(stream, A0, 1024, P(base + 20), 1024, P(base + 21), A4, A4, 512,
          Bc * Tc, 512, 1024, ACT_NONE);
  }

  // ---- coordinate attention ------------------------------------------------
  {
    int cb = 69, mdim = 16;
    k_ycat_copy<<<NBLK(SZ1), 256, 0, stream>>>(A4, A3, Bc, Tc, Fc);
    k_mean_t<<<NBLK(Bc * Fc), 256, 0, stream>>>(A4, A3, Bc, Tc, Fc);
    gemm1(stream, A3, 512, P(cb + 0), 512, P(cb + 1), 0, y1, mdim,
          Bc * 513, mdim, 512, ACT_NONE);
    k_bn_hswish<<<NBLK(Bc * 513 * mdim), 256, 0, stream>>>(
        y1, P(cb + 4), P(cb + 5), P(cb + 2), P(cb + 3), Bc * 513 * mdim, mdim);
    gemm(stream, y1, mdim, P(cb + 6), mdim, 0, P(cb + 7), 0, A5, 512,
         512, 512, mdim, 1.0f, ACT_SIGMOID, Bc, 1,
         (long long)513 * mdim, 0, 0, 0, (long long)Tc * Fc, 0);
    gemm(stream, y1 + 512 * mdim, mdim, P(cb + 8), mdim, 0, P(cb + 9), 0,
         awb, 512, 1, 512, mdim, 1.0f, ACT_SIGMOID, Bc, 1,
         (long long)513 * mdim, 0, 0, 0, (long long)Fc, 0);
    k_coord_mul<<<NBLK(SZ1), 256, 0, stream>>>(A4, A5, awb, A7, Bc, Tc, Fc);
  }

  // ---- trend attention (S=1, 8 heads x 64) ---------------------------------
  {
    int tb = 79;
    int tilesM = Tc / 16, tilesN = Fc / 16;
    dim3 gcv((unsigned)((tilesM * tilesN + 7) / 8), 1, (unsigned)Bc);
    float* Qb = A2; float* Kb = A2 + SZ1; float* Vb = A5; float* Ob = A6;
    k_conv_chlast<<<gcv, 256, 0, stream>>>(A7, P(tb + 0), P(tb + 1), A3, Tc, tilesM, tilesN);
    gemm1(stream, A3, 512, P(tb + 4), 512, P(tb + 5), 0, Qb, 512, Bc * Tc, 512, 512, ACT_NONE);
    k_conv_chlast<<<gcv, 256, 0, stream>>>(A7, P(tb + 2), P(tb + 3), A3, Tc, tilesM, tilesN);
    gemm1(stream, A3, 512, P(tb + 6), 512, P(tb + 7), 0, Kb, 512, Bc * Tc, 512, 512, ACT_NONE);
    gemm1(stream, A7, 512, P(tb + 8), 512, P(tb + 9), 0, Vb, 512, Bc * Tc, 512, 512, ACT_NONE);
    for (int b = 0; b < Bc; ++b) {
      const float* Qp = Qb + (long long)b * Tc * Fc;
      const float* Kp = Kb + (long long)b * Tc * Fc;
      const float* Vp = Vb + (long long)b * Tc * Fc;
      float* Op = Ob + (long long)b * Tc * Fc;
      gemm(stream, Qp, 512, Kp, 512, 0, 0, 0, scores, 512, 512, 512, 64,
           0.125f, ACT_NONE, 8, 8, 0, 64, 0, 64, 0, (long long)Tc * Tc);
      k_softmax<<<8 * Tc, 256, 0, stream>>>(scores, Tc);
      gemm(stream, scores, 512, Vp, 512, 1, 0, 0, Op, 512, 512, 64, 512,
           1.0f, ACT_NONE, 8, 8, 0, (long long)Tc * Tc, 0, 64, 0, 64);
    }
    gemm1(stream, Ob, 512, P(tb + 10), 512, P(tb + 11), A7, A3, 512,
          Bc * Tc, 512, 512, ACT_NONE);
    k_layernorm<<<Bc * Tc, 256, 0, stream>>>(A3, P(tb + 12), P(tb + 13), A3, 512);
  }

  // ---- final FC: (16, 262144) @ fc_w^T + fc_b -> d_out (16,3) --------------
  gemm1(stream, A3, Tc * Fc, P(93), Tc * Fc, P(94), 0,
        (float*)d_out, 3, Bc, 3, Tc * Fc, ACT_NONE);
  #undef P
}
